// basicEmotionPredictions_48799418417238
// MI455X (gfx1250) — compile-verified, matrix operand order checked by reading, not run on hardware
//
#include <hip/hip_runtime.h>
#include <hip/hip_bf16.h>

typedef __attribute__((ext_vector_type(16))) _Float16 v16h;
typedef __attribute__((ext_vector_type(8)))  float    v8f;

#define B_N      32768
#define F_IN     64
#define HID      32
#define E_LEN    25
#define NCOL     2400     // 96 heads * 25
#define BN_EPS   1e-5f
#define L1_EPS   1e-12f

// Branchless SELU: exp is evaluated unconditionally on min(x,0) (bounded, no
// overflow, exact on the negative arm), final pick is a v_cndmask select.
__device__ __forceinline__ float selu_f(float x) {
  const float alpha = 1.6732632423543772f;
  const float scale = 1.0507009873554805f;
  const float neg = scale * alpha * (__expf(fminf(x, 0.f)) - 1.f);
  const float pos = scale * x;
  return (x > 0.f) ? pos : neg;
}

// One butterfly step of a 16-lane-row sum using DPP (no LDS, no dscnt waits).
template <int CTRL>
__device__ __forceinline__ float dpp_add(float x) {
  int yi = __builtin_amdgcn_update_dpp(0, __float_as_int(x), CTRL,
                                       /*row_mask=*/0xF, /*bank_mask=*/0xF,
                                       /*bound_ctrl=*/true);
  return x + __int_as_float(yi);
}

// Sum across each 16-lane half independently (all lanes get their half's sum).
__device__ __forceinline__ float row16_sum(float s) {
  s = dpp_add<0xB1>(s);   // quad_perm [1,0,3,2]  : xor 1
  s = dpp_add<0x4E>(s);   // quad_perm [2,3,0,1]  : xor 2
  s = dpp_add<0x141>(s);  // row_half_mirror (i^7): merge quads within 8
  s = dpp_add<0x140>(s);  // row_mirror     (i^15): merge 8-groups within 16
  return s;
}

// ---------------- Kernel 1: h2 = selu(X@W1+b1)@W2+b2, per-block batch stats ---
__global__ __launch_bounds__(256)
void k1_features(const float* __restrict__ X,
                 const float* __restrict__ W1, const float* __restrict__ b1,
                 const float* __restrict__ W2, const float* __restrict__ b2,
                 float* __restrict__ h2g, float* __restrict__ partials) {
  __shared__ float w1s[F_IN * HID];
  __shared__ float w2s[HID * HID];
  __shared__ float b1s[HID], b2s[HID];
  __shared__ float ssum[HID], ssq[HID];
  const int tid = threadIdx.x;
  for (int i = tid; i < F_IN * HID; i += 256) w1s[i] = W1[i];
  for (int i = tid; i < HID * HID; i += 256) w2s[i] = W2[i];
  if (tid < HID) { b1s[tid] = b1[tid]; b2s[tid] = b2[tid]; ssum[tid] = 0.f; ssq[tid] = 0.f; }
  __syncthreads();

  const int b = blockIdx.x * 256 + tid;
  const float4* xv = (const float4*)(X + (size_t)b * F_IN);

  float h[HID];
  #pragma unroll
  for (int f = 0; f < HID; ++f) h[f] = b1s[f];
  for (int kk = 0; kk < F_IN / 4; ++kk) {
    float4 t = xv[kk];
    float xs4[4] = {t.x, t.y, t.z, t.w};
    #pragma unroll
    for (int i = 0; i < 4; ++i) {
      float xk = xs4[i];
      #pragma unroll
      for (int f = 0; f < HID; ++f) h[f] = fmaf(xk, w1s[(kk * 4 + i) * HID + f], h[f]);
    }
  }
  #pragma unroll
  for (int f = 0; f < HID; ++f) h[f] = selu_f(h[f]);

  const int lane = tid & 31;
  for (int f2 = 0; f2 < HID; ++f2) {
    float acc = b2s[f2];
    #pragma unroll
    for (int f = 0; f < HID; ++f) acc = fmaf(h[f], w2s[f * HID + f2], acc);
    h2g[(size_t)b * HID + f2] = acc;
    float s = acc, q = acc * acc;
    #pragma unroll
    for (int m = 1; m < 32; m <<= 1) { s += __shfl_xor(s, m, 32); q += __shfl_xor(q, m, 32); }
    if (lane == 0) { atomicAdd(&ssum[f2], s); atomicAdd(&ssq[f2], q); }
  }
  __syncthreads();
  if (tid < HID) {
    partials[blockIdx.x * 64 + tid]      = ssum[tid];
    partials[blockIdx.x * 64 + 32 + tid] = ssq[tid];
  }
}

// ---------------- Kernel 2: fold 128 partials into fused BN scale/shift -------
__global__ void k2_stats(const float* __restrict__ partials,
                         const float* __restrict__ gamma, const float* __restrict__ beta,
                         float* __restrict__ sc_sh) {
  const int f = threadIdx.x;  // 32 threads
  float s = 0.f, q = 0.f;
  for (int i = 0; i < 128; ++i) { s += partials[i * 64 + f]; q += partials[i * 64 + 32 + f]; }
  const float mu  = s / (float)B_N;
  const float var = q / (float)B_N - mu * mu;
  const float sc  = gamma[f] * rsqrtf(var + BN_EPS);
  sc_sh[f]      = sc;
  sc_sh[32 + f] = beta[f] - mu * sc;
}

// ---------------- Kernel 3: e = f16(selu(h2*scale+shift)), [B,32] row-major ---
__global__ __launch_bounds__(256)
void k3_normcvt(const float* __restrict__ h2g, const float* __restrict__ sc_sh,
                unsigned int* __restrict__ e16) {
  const int idx = blockIdx.x * 256 + threadIdx.x;   // B_N*HID/2 float2's
  const int f = (idx * 2) & (HID - 1);
  const float2 hv = ((const float2*)h2g)[idx];
  const float v0 = selu_f(hv.x * sc_sh[f]     + sc_sh[32 + f]);
  const float v1 = selu_f(hv.y * sc_sh[f + 1] + sc_sh[32 + f + 1]);
  union { _Float16 h[2]; unsigned int u; } p;
  p.h[0] = (_Float16)v0; p.h[1] = (_Float16)v1;
  e16[idx] = p.u;
}

// ---------------- Kernel 4: per-head WMMA GEMM + SELU + L1 normalize ----------
// wave w of 8 owns head hg = blockIdx.y*8+w; B fragments loaded once, reused
// across 16 M-tiles (software-pipelined one tile deep). 2 WMMAs per tile cover
// E padded 25 -> 32. Epilogue is fully branchless: the tail store degenerates
// to a same-address/same-value rewrite for lanes n>=9.
__global__ __launch_bounds__(256)
void k4_heads(const _Float16* __restrict__ e16,
              const float* __restrict__ Wh, const float* __restrict__ bh,
              float* __restrict__ out) {
  const int tid  = threadIdx.x;
  const int w    = tid >> 5;
  const int lane = tid & 31;
  const int n    = lane & 15;
  const int hi   = lane >> 4;
  const bool tl  = (n < 9);              // tail columns 16..24 valid
  const int hg   = blockIdx.y * 8 + w;
  const float* whh = Wh + (size_t)hg * (HID * E_LEN);
  const int e1 = tl ? (16 + n) : 0;      // clamped index, load unconditional

  // B fragments: lane n holds column n; half j holds K = j + (j>=8?8:0) + hi*8
  v16h bf0, bf1;
  #pragma unroll
  for (int j = 0; j < 16; ++j) {
    const int k = j + ((j >> 3) << 3) + (hi << 3);
    const float w0 = whh[k * E_LEN + n];
    const float w1 = whh[k * E_LEN + e1];
    bf0[j] = (_Float16)w0;
    bf1[j] = (_Float16)(tl ? w1 : 0.f);
  }
  const float bias0 = bh[hg * E_LEN + n];
  const float bias1 = bh[hg * E_LEN + e1];  // garbage for !tl, masked below

  // A-fragment: lanes L and L+16 both hold row m0+(L&15); K-halves live at
  // byte offsets hi*16 and 32+hi*16 of the 64B e16 row.
  auto loadA = [&](int mt) -> v16h {
    const int row = mt * 16 + n;
    const char* eb = (const char*)e16 + (size_t)row * (HID * 2);
    union { v16h v; uint4 u[2]; } af;
    af.u[0] = *(const uint4*)(eb + hi * 16);
    af.u[1] = *(const uint4*)(eb + 32 + hi * 16);
    return af.v;
  };

  // Branchless epilogue for one 16-row tile.
  auto epilogue = [&](int m0, const v8f& c0, const v8f& c1) {
    float* bp  = out + (size_t)(m0 + hi * 8) * NCOL + hg * E_LEN + n;
    float* bp2 = bp + (tl ? 16 : 0);     // invalid lanes alias the first store
    #pragma unroll
    for (int r = 0; r < 8; ++r) {
      const float v0  = selu_f(c0[r] + bias0);
      const float v1s = selu_f(c1[r] + bias1);
      const float v1  = tl ? v1s : 0.f;                  // cndmask, no branch
      const float s   = row16_sum(fabsf(v0) + fabsf(v1)); // DPP butterfly
      const float rs  = __builtin_amdgcn_rcpf(fmaxf(s, L1_EPS)); // v_rcp_f32
      bp [(size_t)r * NCOL] = v0 * rs;                   // sign(d)*|d|/l1 == d/l1
      bp2[(size_t)r * NCOL] = (tl ? v1s : v0) * rs;      // tail or benign rewrite
    }
  };

  const int mt0 = blockIdx.x * 16;
  v16h a = loadA(mt0);
  v8f c0 = __builtin_amdgcn_wmma_f32_16x16x32_f16(false, a, false, bf0,
                                                  (short)0, (v8f){}, false, false);
  v8f c1 = __builtin_amdgcn_wmma_f32_16x16x32_f16(false, a, false, bf1,
                                                  (short)0, (v8f){}, false, false);
  for (int i = 0; i < 15; ++i) {
    const v16h an = loadA(mt0 + i + 1);
    const v8f n0 = __builtin_amdgcn_wmma_f32_16x16x32_f16(false, an, false, bf0,
                                                          (short)0, (v8f){}, false, false);
    const v8f n1 = __builtin_amdgcn_wmma_f32_16x16x32_f16(false, an, false, bf1,
                                                          (short)0, (v8f){}, false, false);
    epilogue((mt0 + i) * 16, c0, c1);   // fills the WMMA hazard window
    c0 = n0; c1 = n1;
  }
  epilogue((mt0 + 15) * 16, c0, c1);
}

// ---------------- launcher ----------------------------------------------------
extern "C" void kernel_launch(void* const* d_in, const int* in_sizes, int n_in,
                              void* d_out, int out_size, void* d_ws, size_t ws_size,
                              hipStream_t stream) {
  const float* X     = (const float*)d_in[0];
  const float* W1    = (const float*)d_in[1];
  const float* b1    = (const float*)d_in[2];
  const float* W2    = (const float*)d_in[3];
  const float* b2    = (const float*)d_in[4];
  const float* gamma = (const float*)d_in[5];
  const float* beta  = (const float*)d_in[6];
  const float* Wh    = (const float*)d_in[7];
  const float* bh    = (const float*)d_in[8];
  float* out = (float*)d_out;

  float* ws = (float*)d_ws;
  float*        h2g      = ws;                         // 1,048,576 f32 (4 MB)
  float*        partials = ws + 1048576;               // 8,192 f32
  float*        sc_sh    = ws + 1048576 + 8192;        // 64 f32
  unsigned int* e16u     = (unsigned int*)(ws + 1048576 + 8192 + 64);  // 2 MB f16, 16B aligned
  const _Float16* e16    = (const _Float16*)e16u;

  k1_features<<<dim3(B_N / 256), dim3(256), 0, stream>>>(X, W1, b1, W2, b2, h2g, partials);
  k2_stats   <<<dim3(1),         dim3(32),  0, stream>>>(partials, gamma, beta, sc_sh);
  k3_normcvt <<<dim3(B_N * HID / 512), dim3(256), 0, stream>>>(h2g, sc_sh, e16u);
  k4_heads   <<<dim3(B_N / 256, 12), dim3(256), 0, stream>>>(e16, Wh, bh, out);
}
